// DecoderLayer_83554293776404
// MI455X (gfx1250) — matrix operand order repaired and checked by reading, not compiled, not run on hardware
//
#include <hip/hip_runtime.h>
#include <stdint.h>

#define S_    1024
#define D_    1024
#define H_    16
#define KV_   4
#define HD_   64
#define E_    8
#define TOPK_ 2
#define F_    1024
#define CAP_  2560   // 2048 assignments + per-expert pad to 64, multiple of 128

typedef __attribute__((ext_vector_type(16))) __bf16          v16bf;
typedef __attribute__((ext_vector_type(16))) unsigned short  v16us;
typedef __attribute__((ext_vector_type(8)))  float           v8f;

static __device__ __forceinline__ unsigned short f2bf(float f) {
  union { float f; uint32_t u; } v; v.f = f;
  uint32_t u = v.u;
  u += 0x7fffu + ((u >> 16) & 1u);          // round-to-nearest-even
  return (unsigned short)(u >> 16);
}
static __device__ __forceinline__ float bf2f(unsigned short b) {
  union { uint32_t u; float f; } v; v.u = ((uint32_t)b) << 16;
  return v.f;
}

static __device__ __forceinline__ v8f wmma_bf16(v16us a, v16us b, v8f c) {
  return __builtin_amdgcn_wmma_f32_16x16x32_bf16(
      false, __builtin_bit_cast(v16bf, a),
      false, __builtin_bit_cast(v16bf, b),
      (short)0, c, false, false);
}

// A-operand (16x32 bf16): lane half g holds K = 8g..8g+7 and 16+8g..16+8g+7
static __device__ __forceinline__ v16us load_frag_a(const unsigned short* rowptr, int k0, int g) {
  v16us r;
  ((uint4*)&r)[0] = *(const uint4*)(rowptr + k0 + 8 * g);
  ((uint4*)&r)[1] = *(const uint4*)(rowptr + k0 + 16 + 8 * g);
  return r;
}
// B-operand (32x16 bf16) fed from B^T (N x K row-major): lane half g holds K = 8g..8g+7 and 16+8g..16+8g+7
static __device__ __forceinline__ v16us load_frag_b(const unsigned short* rowptr, int k0, int g) {
  const unsigned short* p = rowptr + k0 + 8 * g;
  v16us r;
  ((uint4*)&r)[0] = *(const uint4*)(p);
  ((uint4*)&r)[1] = *(const uint4*)(p + 16);
  return r;
}

// ---------------------------------------------------------------------------
// Generic bf16 WMMA GEMM.  A: M x K row-major (optional row gather).
// B passed as B^T: N x K row-major (optional per-64-row expert offset).
// Wave tile 64x64 (4x4 16x16 frags, 16 wmma / 16 b128 loads per K-step);
// block = 8 waves (2 along M, 4 along N) -> 128x256 tile.
// mode 0: Cf = acc ; mode 1: Cf = Aux + acc ; mode 2: Cb = bf16(silu(Aux)*acc)
// ---------------------------------------------------------------------------
__global__ __launch_bounds__(256) void gemm_bf16_wmma(
    const unsigned short* __restrict__ A, int lda,
    const unsigned short* __restrict__ B, int ldb,
    const int* __restrict__ gather,
    const int* __restrict__ exprow, long long bstride,
    float* __restrict__ Cf, const float* __restrict__ Aux,
    unsigned short* __restrict__ Cb,
    int M, int N, int K, int ldc, int mode)
{
  const int lane = threadIdx.x & 31;
  const int wave = threadIdx.x >> 5;
  const int g    = lane >> 4;
  const int hl   = lane & 15;
  const int wm   = wave & 1;          // 2 waves along M
  const int wn   = wave >> 1;         // 4 waves along N
  const int row0 = blockIdx.y * 128 + wm * 64;
  const int col0 = blockIdx.x * 256 + wn * 64;

  const unsigned short* Bp = B;
  if (exprow) Bp += (long long)exprow[row0] * bstride;

  const unsigned short* arow[4];
#pragma unroll
  for (int i = 0; i < 4; ++i) {
    int r = row0 + i * 16 + hl;
    if (gather) { int t = gather[r]; if (t < 0) t = 0; r = t; }
    arow[i] = A + (long long)r * lda;
  }
  const unsigned short* brow[4];
#pragma unroll
  for (int j = 0; j < 4; ++j)
    brow[j] = Bp + (long long)(col0 + j * 16 + hl) * ldb;

  v8f acc[4][4] = {};

  for (int k0 = 0; k0 < K; k0 += 32) {
    v16us af[4], bfr[4];
#pragma unroll
    for (int i = 0; i < 4; ++i) af[i] = load_frag_a(arow[i], k0, g);
#pragma unroll
    for (int j = 0; j < 4; ++j) bfr[j] = load_frag_b(brow[j], k0, g);
#pragma unroll
    for (int i = 0; i < 4; ++i) {
      __builtin_prefetch(arow[i] + k0 + 128, 0, 3);   // near-cache prefetch
#pragma unroll
      for (int j = 0; j < 4; ++j)
        acc[i][j] = wmma_bf16(af[i], bfr[j], acc[i][j]);
    }
  }

#pragma unroll
  for (int i = 0; i < 4; ++i)
#pragma unroll
    for (int j = 0; j < 4; ++j)
#pragma unroll
      for (int e = 0; e < 8; ++e) {
        int r = row0 + i * 16 + e + 8 * g;   // C layout: VGPR e, lane half g
        int c = col0 + j * 16 + hl;
        if (r >= M || c >= N) continue;
        long long o = (long long)r * ldc + c;
        float v = acc[i][j][e];
        if (mode == 0) {
          Cf[o] = v;
        } else if (mode == 1) {
          Cf[o] = Aux[o] + v;
        } else {
          float gv = Aux[o];
          float sg = gv / (1.0f + __expf(-gv));   // silu
          Cb[o] = f2bf(sg * v);
        }
      }
}

// ---------------------------------------------------------------------------
// RMSNorm (f32 in) -> bf16 out, one block per row
// ---------------------------------------------------------------------------
__global__ __launch_bounds__(256) void rmsnorm_bf16_kernel(
    const float* __restrict__ X, const float* __restrict__ W,
    unsigned short* __restrict__ Y, int ncols)
{
  const int row = blockIdx.x;
  const float* x = X + (long long)row * ncols;
  float ss = 0.f;
  for (int c = threadIdx.x; c < ncols; c += 256) { float v = x[c]; ss += v * v; }
#pragma unroll
  for (int d = 1; d < 32; d <<= 1) ss += __shfl_xor(ss, d, 32);
  __shared__ float red[8];
  if ((threadIdx.x & 31) == 0) red[threadIdx.x >> 5] = ss;
  __syncthreads();
  ss = 0.f;
#pragma unroll
  for (int w = 0; w < 8; ++w) ss += red[w];
  float r = rsqrtf(ss / (float)ncols + 1e-5f);
  for (int c = threadIdx.x; c < ncols; c += 256)
    Y[(long long)row * ncols + c] = f2bf(x[c] * r * W[c]);
}

// ---------------------------------------------------------------------------
// RMSNorm + RoPE + head-major relayout: out[(head*S + s)*HD + d] bf16
// ---------------------------------------------------------------------------
__global__ __launch_bounds__(256) void norm_rope_kernel(
    const float* __restrict__ P, const float* __restrict__ W,
    const float* __restrict__ Cos, const float* __restrict__ Sin,
    unsigned short* __restrict__ Out, int nheads)
{
  const int s = blockIdx.x;
  const int ncols = nheads * HD_;
  const float* x = P + (long long)s * ncols;
  float ss = 0.f;
  for (int c = threadIdx.x; c < ncols; c += 256) { float v = x[c]; ss += v * v; }
#pragma unroll
  for (int d = 1; d < 32; d <<= 1) ss += __shfl_xor(ss, d, 32);
  __shared__ float red[8];
  if ((threadIdx.x & 31) == 0) red[threadIdx.x >> 5] = ss;
  __syncthreads();
  ss = 0.f;
#pragma unroll
  for (int w = 0; w < 8; ++w) ss += red[w];
  float r = rsqrtf(ss / (float)ncols + 1e-5f);
  for (int c = threadIdx.x; c < ncols; c += 256) {
    int h = c >> 6, d = c & 63;
    float a = x[c] * r * W[c];
    int pc = (d < 32) ? (c + 32) : (c - 32);
    float b = x[pc] * r * W[pc];
    if (d < 32) b = -b;                       // rotate_half
    float o = a * Cos[s * HD_ + d] + b * Sin[s * HD_ + d];
    Out[((long long)h * S_ + s) * HD_ + d] = f2bf(o);
  }
}

// V (S x KV*HD f32) -> Vt[kv][d][s] bf16  (B^T layout for the PV gemm)
__global__ __launch_bounds__(256) void vtrans_kernel(
    const float* __restrict__ vp, unsigned short* __restrict__ vt)
{
  int i = blockIdx.x * 256 + threadIdx.x;       // S*KV*HD elements
  int s = i >> 8;
  int c = i & 255;
  int kvh = c >> 6, d = c & 63;
  vt[((long long)kvh * HD_ + d) * S_ + s] = f2bf(vp[i]);
}

// ---------------------------------------------------------------------------
// Flash attention: one wave per (head, 16-query tile), causal, online softmax
// ---------------------------------------------------------------------------
__global__ __launch_bounds__(128) void attn_kernel(
    const unsigned short* __restrict__ Q,   // [H][S][HD]
    const unsigned short* __restrict__ Kt,  // [KV][S][HD]  (= B^T for scores)
    const unsigned short* __restrict__ Vt,  // [KV][HD][S]  (= B^T for PV)
    unsigned short* __restrict__ O)         // [S][D] bf16
{
  __shared__ unsigned short Pl[4][16 * 32];
  const int lane = threadIdx.x & 31;
  const int wave = threadIdx.x >> 5;
  const int g = lane >> 4, hl = lane & 15;
  const int task = blockIdx.x * 4 + wave;
  const int h  = task >> 6;                  // 64 q-tiles per head
  const int qt = task & 63;
  const int kv = h >> 2;                     // GQA: 4 q-heads per kv-head

  const unsigned short* qrow = Q + ((long long)h * S_ + qt * 16 + hl) * HD_;
  v16us qf[2];
#pragma unroll
  for (int t = 0; t < 2; ++t) qf[t] = load_frag_a(qrow, t * 32, g);

  v8f acc[4] = {};
  float rm[8], rs[8];
#pragma unroll
  for (int e = 0; e < 8; ++e) { rm[e] = -3.0e38f; rs[e] = 0.f; }

  const int nkb = (qt * 16 + 15) / 32 + 1;   // causal: only needed key blocks
  for (int kb = 0; kb < nkb; ++kb) {
    v8f sc[2];
#pragma unroll
    for (int t2 = 0; t2 < 2; ++t2) {
      const unsigned short* krow =
          Kt + ((long long)kv * S_ + kb * 32 + t2 * 16 + hl) * HD_;
      v8f sv = {};
#pragma unroll
      for (int kk = 0; kk < 2; ++kk)
        sv = wmma_bf16(qf[kk], load_frag_b(krow, kk * 32, g), sv);
      sc[t2] = sv;
    }
#pragma unroll
    for (int e = 0; e < 8; ++e) {
      int qi = qt * 16 + e + 8 * g;
#pragma unroll
      for (int t2 = 0; t2 < 2; ++t2) {
        int kj = kb * 32 + t2 * 16 + hl;
        float v = sc[t2][e] * 0.125f;        // HD^-0.5
        sc[t2][e] = (kj > qi) ? -3.0e38f : v;
      }
      float m = fmaxf(sc[0][e], sc[1][e]);
#pragma unroll
      for (int d = 1; d < 16; d <<= 1) m = fmaxf(m, __shfl_xor(m, d, 32));
      float nm = fmaxf(rm[e], m);
      float alpha = __expf(rm[e] - nm);
      float psum = 0.f;
#pragma unroll
      for (int t2 = 0; t2 < 2; ++t2) {
        float pv = __expf(sc[t2][e] - nm);
        sc[t2][e] = pv;
        psum += pv;
      }
#pragma unroll
      for (int d = 1; d < 16; d <<= 1) psum += __shfl_xor(psum, d, 32);
      rm[e] = nm;
      rs[e] = rs[e] * alpha + psum;
#pragma unroll
      for (int t = 0; t < 4; ++t) acc[t][e] *= alpha;
    }
    // relay P (C layout) -> A-operand layout through per-wave LDS slab
    unsigned short* L = Pl[wave];
#pragma unroll
    for (int e = 0; e < 8; ++e)
#pragma unroll
      for (int t2 = 0; t2 < 2; ++t2)
        L[(e + 8 * g) * 32 + t2 * 16 + hl] = f2bf(sc[t2][e]);
    v16us pf = load_frag_a(L + hl * 32, 0, g);
#pragma unroll
    for (int t = 0; t < 4; ++t) {
      const unsigned short* vr =
          Vt + ((long long)kv * HD_ + t * 16 + hl) * S_ + kb * 32;
      acc[t] = wmma_bf16(pf, load_frag_b(vr, 0, g), acc[t]);
    }
  }
#pragma unroll
  for (int t = 0; t < 4; ++t)
#pragma unroll
    for (int e = 0; e < 8; ++e) {
      int srow = qt * 16 + e + 8 * g;
      O[(long long)srow * D_ + h * HD_ + t * 16 + hl] = f2bf(acc[t][e] / rs[e]);
    }
}

// ---------------------------------------------------------------------------
// MoE routing
// ---------------------------------------------------------------------------
__global__ __launch_bounds__(256) void route1_kernel(
    const unsigned short* __restrict__ H2, const float* __restrict__ Wg,
    float* __restrict__ probs, int* __restrict__ choices, int* __restrict__ counts)
{
  const int s = blockIdx.x;
  float part[E_];
#pragma unroll
  for (int e = 0; e < E_; ++e) part[e] = 0.f;
  for (int c = threadIdx.x; c < D_; c += 256) {
    float hv = bf2f(H2[(long long)s * D_ + c]);
#pragma unroll
    for (int e = 0; e < E_; ++e) part[e] += hv * Wg[c * E_ + e];
  }
#pragma unroll
  for (int e = 0; e < E_; ++e)
#pragma unroll
    for (int d = 1; d < 32; d <<= 1) part[e] += __shfl_xor(part[e], d, 32);
  __shared__ float red[8][E_];
  if ((threadIdx.x & 31) == 0)
#pragma unroll
    for (int e = 0; e < E_; ++e) red[threadIdx.x >> 5][e] = part[e];
  __syncthreads();
  if (threadIdx.x == 0) {
    float lg[E_], pe[E_];
#pragma unroll
    for (int e = 0; e < E_; ++e) {
      float a = 0.f;
      for (int w = 0; w < 8; ++w) a += red[w][e];
      lg[e] = a;
    }
    float m = lg[0];
    for (int e = 1; e < E_; ++e) m = fmaxf(m, lg[e]);
    float sum = 0.f;
    for (int e = 0; e < E_; ++e) { pe[e] = __expf(lg[e] - m); sum += pe[e]; }
    for (int e = 0; e < E_; ++e) pe[e] /= sum;
    int e0 = 0;
    for (int e = 1; e < E_; ++e) if (pe[e] > pe[e0]) e0 = e;
    int e1 = (e0 == 0) ? 1 : 0;
    for (int e = 0; e < E_; ++e) if (e != e0 && pe[e] > pe[e1]) e1 = e;
    probs[s * 2] = pe[e0]; probs[s * 2 + 1] = pe[e1];
    choices[s * 2] = e0;   choices[s * 2 + 1] = e1;
    atomicAdd(&counts[e0], 1);
    atomicAdd(&counts[e1], 1);
  }
}

__global__ __launch_bounds__(256) void route2_kernel(
    const int* __restrict__ counts, int* __restrict__ base, int* __restrict__ cursor,
    int* __restrict__ tok, int* __restrict__ exprow)
{
  __shared__ int sb[E_ + 1];
  if (threadIdx.x == 0) {
    int cum = 0;
    for (int e = 0; e < E_; ++e) {
      sb[e] = cum; base[e] = cum; cursor[e] = 0;
      cum += (counts[e] + 63) & ~63;          // 64-align so each wave tile is one expert
    }
    sb[E_] = cum;
  }
  __syncthreads();
  for (int r = threadIdx.x; r < CAP_; r += 256) {
    tok[r] = -1;
    int e = 0;
    for (int ee = 1; ee < E_; ++ee) if (r >= sb[ee]) e = ee;
    exprow[r] = e;
  }
}

__global__ __launch_bounds__(256) void route3_kernel(
    const int* __restrict__ choices, const int* __restrict__ base,
    int* __restrict__ cursor, int* __restrict__ tok, int* __restrict__ slotm)
{
  int i = blockIdx.x * 256 + threadIdx.x;
  if (i >= S_ * TOPK_) return;
  int e = choices[i];
  int pos = atomicAdd(&cursor[e], 1);
  int row = base[e] + pos;
  tok[row] = i >> 1;
  slotm[i] = row;
}

__global__ __launch_bounds__(256) void combine_kernel(
    const float* __restrict__ down, const float* __restrict__ probs,
    const int* __restrict__ slotm, float* __restrict__ out)
{
  int i = blockIdx.x * 256 + threadIdx.x;     // over S*D
  int s = i >> 10;
  int d = i & (D_ - 1);
  float v = out[i];
  v += probs[s * 2]     * down[(long long)slotm[s * 2]     * D_ + d];
  v += probs[s * 2 + 1] * down[(long long)slotm[s * 2 + 1] * D_ + d];
  out[i] = v;
}

// ---------------------------------------------------------------------------
// fp32 -> bf16 conversions (plain, and with transpose for the N x K weights)
// ---------------------------------------------------------------------------
__global__ __launch_bounds__(256) void cvt_bf16_kernel(
    const float* __restrict__ in, unsigned short* __restrict__ o, long long n)
{
  long long i = (long long)blockIdx.x * 256 + threadIdx.x;
  if (i < n) o[i] = f2bf(in[i]);
}
__global__ __launch_bounds__(256) void cvt_bf16_t_kernel(
    const float* __restrict__ in, unsigned short* __restrict__ o, int R, int C)
{
  long long i = (long long)blockIdx.x * 256 + threadIdx.x;
  if (i < (long long)R * C) {
    int r = (int)(i / C), c = (int)(i % C);
    o[(long long)c * R + r] = f2bf(in[i]);    // out is C x R (= W^T)
  }
}
__global__ void init_counts_kernel(int* counts) {
  if (threadIdx.x < E_) counts[threadIdx.x] = 0;
}

// ---------------------------------------------------------------------------
extern "C" void kernel_launch(void* const* d_in, const int* in_sizes, int n_in,
                              void* d_out, int out_size, void* d_ws, size_t ws_size,
                              hipStream_t stream)
{
  (void)in_sizes; (void)n_in; (void)out_size; (void)ws_size;
  const float* x      = (const float*)d_in[0];
  const float* cosb   = (const float*)d_in[1];
  const float* sinb   = (const float*)d_in[2];
  // d_in[3] = mask (causal, handled analytically), d_in[4] = layer_idx (unused)
  const float* w_in   = (const float*)d_in[5];
  const float* wq     = (const float*)d_in[6];
  const float* wk     = (const float*)d_in[7];
  const float* wv     = (const float*)d_in[8];
  const float* wo     = (const float*)d_in[9];
  const float* w_qn   = (const float*)d_in[10];
  const float* w_kn   = (const float*)d_in[11];
  const float* w_pn   = (const float*)d_in[12];
  const float* w_gate = (const float*)d_in[13];
  const float* up_p   = (const float*)d_in[14];
  const float* gate_p = (const float*)d_in[15];
  const float* down_p = (const float*)d_in[16];
  float* out = (float*)d_out;

  char* p = (char*)d_ws;
  auto alloc = [&](size_t bytes) {
    char* r = p;
    p += (bytes + 255) & ~(size_t)255;
    return r;
  };
  unsigned short* h_bf   = (unsigned short*)alloc(2ll * S_ * D_);
  unsigned short* wqT    = (unsigned short*)alloc(2ll * D_ * D_);
  unsigned short* wkT    = (unsigned short*)alloc(2ll * 256 * D_);
  unsigned short* wvT    = (unsigned short*)alloc(2ll * 256 * D_);
  unsigned short* woT    = (unsigned short*)alloc(2ll * D_ * D_);
  unsigned short* upB    = (unsigned short*)alloc(2ll * E_ * F_ * D_);
  unsigned short* gateB  = (unsigned short*)alloc(2ll * E_ * F_ * D_);
  unsigned short* downB  = (unsigned short*)alloc(2ll * E_ * D_ * F_);
  float*          qpre   = (float*)alloc(4ll * S_ * D_);
  float*          kpre   = (float*)alloc(4ll * S_ * 256);
  float*          vpre   = (float*)alloc(4ll * S_ * 256);
  unsigned short* q_bf   = (unsigned short*)alloc(2ll * S_ * D_);
  unsigned short* k_bf   = (unsigned short*)alloc(2ll * KV_ * S_ * HD_);
  unsigned short* v_t    = (unsigned short*)alloc(2ll * KV_ * HD_ * S_);
  unsigned short* attnbf = (unsigned short*)alloc(2ll * S_ * D_);
  unsigned short* h2_bf  = (unsigned short*)alloc(2ll * S_ * D_);
  float*          probs  = (float*)alloc(4ll * S_ * TOPK_);
  int*            choices= (int*)alloc(4ll * S_ * TOPK_);
  int*            counts = (int*)alloc(4ll * E_);
  int*            base   = (int*)alloc(4ll * E_);
  int*            cursor = (int*)alloc(4ll * E_);
  int*            tok    = (int*)alloc(4ll * CAP_);
  int*            exprow = (int*)alloc(4ll * CAP_);
  int*            slotm  = (int*)alloc(4ll * S_ * TOPK_);
  float*          gateBuf= (float*)alloc(4ll * CAP_ * F_);
  unsigned short* hidBf  = (unsigned short*)alloc(2ll * CAP_ * F_);
  float*          downBuf= (float*)alloc(4ll * CAP_ * D_);

  // 1) weights -> bf16 (projections transposed to N x K, MoE already N x K)
  cvt_bf16_t_kernel<<<(D_ * D_) / 256, 256, 0, stream>>>(wq, wqT, D_, D_);
  cvt_bf16_t_kernel<<<(D_ * 256) / 256, 256, 0, stream>>>(wk, wkT, D_, 256);
  cvt_bf16_t_kernel<<<(D_ * 256) / 256, 256, 0, stream>>>(wv, wvT, D_, 256);
  cvt_bf16_t_kernel<<<(D_ * D_) / 256, 256, 0, stream>>>(wo, woT, D_, D_);
  cvt_bf16_kernel<<<(E_ * F_ * D_) / 256, 256, 0, stream>>>(up_p,   upB,   (long long)E_ * F_ * D_);
  cvt_bf16_kernel<<<(E_ * F_ * D_) / 256, 256, 0, stream>>>(gate_p, gateB, (long long)E_ * F_ * D_);
  cvt_bf16_kernel<<<(E_ * D_ * F_) / 256, 256, 0, stream>>>(down_p, downB, (long long)E_ * D_ * F_);

  // 2) input norm, QKV projections
  rmsnorm_bf16_kernel<<<S_, 256, 0, stream>>>(x, w_in, h_bf, D_);
  gemm_bf16_wmma<<<dim3(4, 8), 256, 0, stream>>>(h_bf, D_, wqT, D_, nullptr, nullptr, 0,
                                                 qpre, nullptr, nullptr, S_, D_, D_, D_, 0);
  gemm_bf16_wmma<<<dim3(1, 8), 256, 0, stream>>>(h_bf, D_, wkT, D_, nullptr, nullptr, 0,
                                                 kpre, nullptr, nullptr, S_, 256, D_, 256, 0);
  gemm_bf16_wmma<<<dim3(1, 8), 256, 0, stream>>>(h_bf, D_, wvT, D_, nullptr, nullptr, 0,
                                                 vpre, nullptr, nullptr, S_, 256, D_, 256, 0);

  // 3) q/k norm + RoPE + head layout, v transpose
  norm_rope_kernel<<<S_, 256, 0, stream>>>(qpre, w_qn, cosb, sinb, q_bf, H_);
  norm_rope_kernel<<<S_, 256, 0, stream>>>(kpre, w_kn, cosb, sinb, k_bf, KV_);
  vtrans_kernel<<<(S_ * 256) / 256, 256, 0, stream>>>(vpre, v_t);

  // 4) flash attention (1024 wave-tasks, 4 waves/block)
  attn_kernel<<<256, 128, 0, stream>>>(q_bf, k_bf, v_t, attnbf);

  // 5) output projection + residual -> d_out holds x2
  gemm_bf16_wmma<<<dim3(4, 8), 256, 0, stream>>>(attnbf, D_, woT, D_, nullptr, nullptr, 0,
                                                 out, x, nullptr, S_, D_, D_, D_, 1);

  // 6) post norm + routing
  rmsnorm_bf16_kernel<<<S_, 256, 0, stream>>>(out, w_pn, h2_bf, D_);
  init_counts_kernel<<<1, 32, 0, stream>>>(counts);
  route1_kernel<<<S_, 256, 0, stream>>>(h2_bf, w_gate, probs, choices, counts);
  route2_kernel<<<1, 256, 0, stream>>>(counts, base, cursor, tok, exprow);
  route3_kernel<<<(S_ * TOPK_) / 256, 256, 0, stream>>>(choices, base, cursor, tok, slotm);

  // 7) sparse MoE: grouped GEMMs over gathered token rows
  gemm_bf16_wmma<<<dim3(4, CAP_ / 128), 256, 0, stream>>>(
      h2_bf, D_, gateB, D_, tok, exprow, (long long)F_ * D_,
      gateBuf, nullptr, nullptr, CAP_, F_, D_, F_, 0);
  gemm_bf16_wmma<<<dim3(4, CAP_ / 128), 256, 0, stream>>>(
      h2_bf, D_, upB, D_, tok, exprow, (long long)F_ * D_,
      nullptr, gateBuf, hidBf, CAP_, F_, D_, F_, 2);     // silu(gate)*up -> bf16
  gemm_bf16_wmma<<<dim3(4, CAP_ / 128), 256, 0, stream>>>(
      hidBf, F_, downB, F_, nullptr, exprow, (long long)D_ * F_,
      downBuf, nullptr, nullptr, CAP_, D_, F_, D_, 0);

  // 8) weighted combine into residual output
  combine_kernel<<<(S_ * D_) / 256, 256, 0, stream>>>(downBuf, probs, slotm, out);
}
